// Model_39676907883022
// MI455X (gfx1250) — matrix operand order repaired and checked by reading, not compiled, not run on hardware
//
#include <hip/hip_runtime.h>

typedef __attribute__((ext_vector_type(16))) _Float16 v16h;
typedef __attribute__((ext_vector_type(8)))  _Float16 v8h;
typedef __attribute__((ext_vector_type(2)))  _Float16 v2h;
typedef __attribute__((ext_vector_type(8)))  float    v8f;
typedef __attribute__((ext_vector_type(4)))  float    v4f;

#define B_    16
#define L_    4096
#define D_    64
#define BM    128            // Q rows per block (8 waves x 16)
#define BN    64             // KV rows per iteration
#define NT    (L_ / BN)      // 64 iterations
#define NWAVE 8

static __device__ __forceinline__ v2h pkrtz(float a, float b) {
    return __builtin_bit_cast(v2h, __builtin_amdgcn_cvt_pkrtz(a, b));
}

__launch_bounds__(256, 2)
__global__ void fa_fwd_kernel(const float* __restrict__ Q,
                              const float* __restrict__ K,
                              const float* __restrict__ V,
                              const float* __restrict__ scale_factor,
                              float* __restrict__ O)
{
    // Kt: [buf][kv][d] row-major (contraction d contiguous for S = Q*K^T)
    // Vt: [buf][d][kv-slot] transposed; kv-slot(c) = 32*(c>>5) + 2*(c&15) + ((c>>4)&1)
    // Vones: 16 extra V rows, row 0 = 1.0 -> fused row-sum via a 5th PV tile
    __shared__ __align__(32) _Float16 Kt[2][BN * D_];        // 16 KB
    __shared__ __align__(32) _Float16 Vt[2][D_ * BN];        // 16 KB
    __shared__ __align__(32) _Float16 Vones[16 * BN];        //  2 KB
    __shared__ __align__(32) _Float16 Ps[NWAVE][16 * BN];    // 16 KB

    const int tid  = threadIdx.x;
    const int wave = tid >> 5;
    const int lane = tid & 31;
    const int lo   = lane & 15;
    const int hi   = lane >> 4;

    const int b     = blockIdx.y;
    const int qbase = blockIdx.x * BM + wave * 16;

    // fold softmax scale and log2(e) into Q: scores come out in exp2 domain
    const float qscale = 1.4426950408889634f / scale_factor[0];

    for (int i = tid; i < 16 * BN; i += 256)
        Vones[i] = (_Float16)((i < BN) ? 1.0f : 0.0f);

    // ---- Q as two 16x32 f16 A-fragments, pre-scaled ----
    // A-frag: half h -> k = h + 16*hi
    v16h aq[2];
    {
        const float* qrow = Q + ((size_t)b * L_ + (qbase + lo)) * (size_t)D_;
        #pragma unroll
        for (int h = 0; h < 16; ++h) {
            int d = h + 16 * hi;
            aq[0][h] = (_Float16)(qrow[d] * qscale);
            aq[1][h] = (_Float16)(qrow[32 + d] * qscale);
        }
    }

    // o[0..3]: 16x64 output tiles; o[4]: fused row-sum (l) tile
    v8f o[5];
    #pragma unroll
    for (int t = 0; t < 5; ++t) o[t] = (v8f){};
    float m_r[8];
    #pragma unroll
    for (int r = 0; r < 8; ++r) m_r[r] = -3.0e38f;

    // staging split: 4096 f32 per tile / 256 threads = 16 contiguous each
    const int srow  = tid >> 2;                                   // kv row 0..63
    const int scol  = (tid & 3) * 16;                             // d col start
    const int vslot = 32 * (srow >> 5) + 2 * (srow & 15) + ((srow >> 4) & 1);

    const float* kbase = K + ((size_t)b * L_ + srow) * (size_t)D_ + scol;
    const float* vbase = V + ((size_t)b * L_ + srow) * (size_t)D_ + scol;

    auto load_tile = [&](int kv, v4f kr[4], v4f vr[4]) {
        const float* kq = kbase + (size_t)kv * D_;
        const float* vq = vbase + (size_t)kv * D_;
        #pragma unroll
        for (int j = 0; j < 4; ++j) {
            kr[j] = *(const v4f*)(kq + 4 * j);
            vr[j] = *(const v4f*)(vq + 4 * j);
        }
    };
    auto store_tile = [&](int buf, const v4f kr[4], const v4f vr[4]) {
        union { v8h h[2]; v2h p[8]; } kc, vc;
        #pragma unroll
        for (int j = 0; j < 8; ++j) {
            kc.p[j] = pkrtz(kr[j >> 1][(2 * j) & 3], kr[j >> 1][((2 * j) & 3) + 1]);
            vc.p[j] = pkrtz(vr[j >> 1][(2 * j) & 3], vr[j >> 1][((2 * j) & 3) + 1]);
        }
        *(v8h*)&Kt[buf][srow * D_ + scol]     = kc.h[0];
        *(v8h*)&Kt[buf][srow * D_ + scol + 8] = kc.h[1];
        #pragma unroll
        for (int j = 0; j < 16; ++j)
            Vt[buf][(scol + j) * BN + vslot] = ((const _Float16*)&vc)[j];
    };

    // ---- pipeline prologue: stage tile 0 into buffer 0 ----
    v4f kr[4], vr[4];
    load_tile(0, kr, vr);
    store_tile(0, kr, vr);
    __syncthreads();

    for (int t = 0; t < NT; ++t) {
        const int cur = t & 1;

        // issue next tile's global loads now; consumed after the compute body
        const int kvn = (t + 1 < NT) ? (t + 1) * BN : 0;   // wrap: safe dummy
        load_tile(kvn, kr, vr);
        const int kvp = (t + 2 < NT) ? (t + 2) * BN : 0;
        __builtin_prefetch(kbase + (size_t)kvp * D_, 0, 1);
        __builtin_prefetch(vbase + (size_t)kvp * D_, 0, 1);

        // ---- S = Q*K^T : ks-outer so each accumulator has reuse distance 4 ----
        const _Float16* ktc = &Kt[cur][0];
        v8f s[4];
        #pragma unroll
        for (int nt = 0; nt < 4; ++nt) s[nt] = (v8f){};
        #pragma unroll
        for (int ks = 0; ks < 2; ++ks) {
            #pragma unroll
            for (int nt = 0; nt < 4; ++nt) {
                v16h bk = *(const v16h*)&ktc[(nt * 16 + lo) * D_ + 32 * ks + 16 * hi];
                s[nt] = __builtin_amdgcn_wmma_f32_16x16x32_f16(false, aq[ks], false, bk,
                                                               (short)0, s[nt], false, false);
            }
        }

        // ---- row max: packed f16 butterfly (2 rows per reg) ----
        float mnew[8];
        #pragma unroll
        for (int k = 0; k < 4; ++k) {
            const int r0 = 2 * k, r1 = 2 * k + 1;
            float a0 = fmaxf(fmaxf(s[0][r0], s[1][r0]), fmaxf(s[2][r0], s[3][r0]));
            float a1 = fmaxf(fmaxf(s[0][r1], s[1][r1]), fmaxf(s[2][r1], s[3][r1]));
            union { float f; v2h h; } u, tt;
            u.h = pkrtz(a0, a1);
            #pragma unroll
            for (int msk = 1; msk < 16; msk <<= 1) {
                tt.f = __shfl_xor(u.f, msk, 32);
                u.h  = __builtin_elementwise_max(u.h, tt.h);
            }
            mnew[r0] = fmaxf(m_r[r0], (float)u.h[0]);
            mnew[r1] = fmaxf(m_r[r1], (float)u.h[1]);
        }
        // unconditional rescale: alpha == 1 when max unchanged; no control flow
        #pragma unroll
        for (int r = 0; r < 8; ++r) {
            float alpha = __builtin_amdgcn_exp2f(m_r[r] - mnew[r]);
            o[0][r] *= alpha; o[1][r] *= alpha; o[2][r] *= alpha;
            o[3][r] *= alpha; o[4][r] *= alpha;
            m_r[r] = mnew[r];
        }

        // ---- P = exp2(S - m), packed f16 stores into permuted slots ----
        _Float16* ps = &Ps[wave][0];
        #pragma unroll
        for (int r = 0; r < 8; ++r) {
            float p0 = __builtin_amdgcn_exp2f(s[0][r] - mnew[r]);
            float p1 = __builtin_amdgcn_exp2f(s[1][r] - mnew[r]);
            float p2 = __builtin_amdgcn_exp2f(s[2][r] - mnew[r]);
            float p3 = __builtin_amdgcn_exp2f(s[3][r] - mnew[r]);
            const int row = (r + 8 * hi) * BN;
            *(v2h*)&ps[row + 2 * lo]      = pkrtz(p0, p1);   // slots 2lo, 2lo+1
            *(v2h*)&ps[row + 32 + 2 * lo] = pkrtz(p2, p3);   // slots 32+2lo, +1
        }
        union { v16h v; v8h h[2]; } ap[2];
        ap[0].h[0] = *(const v8h*)&ps[lo * BN + 16 * hi];
        ap[0].h[1] = *(const v8h*)&ps[lo * BN + 8 + 16 * hi];
        ap[1].h[0] = *(const v8h*)&ps[lo * BN + 32 + 16 * hi];
        ap[1].h[1] = *(const v8h*)&ps[lo * BN + 40 + 16 * hi];

        // ---- O += P*V : ks-outer, 5 tiles per run sharing the A operand ----
        const _Float16* vtc = &Vt[cur][0];
        #pragma unroll
        for (int ks = 0; ks < 2; ++ks) {
            #pragma unroll
            for (int dt = 0; dt < 4; ++dt) {
                v16h bv = *(const v16h*)&vtc[(dt * 16 + lo) * BN + 32 * ks + 16 * hi];
                o[dt] = __builtin_amdgcn_wmma_f32_16x16x32_f16(false, ap[ks].v, false, bv,
                                                               (short)0, o[dt], false, false);
            }
            v16h bo = *(const v16h*)&Vones[lo * BN + 32 * ks + 16 * hi];
            o[4] = __builtin_amdgcn_wmma_f32_16x16x32_f16(false, ap[ks].v, false, bo,
                                                          (short)0, o[4], false, false);
        }

        // ---- stage the prefetched tile into the alternate buffer ----
        store_tile(cur ^ 1, kr, vr);
        __syncthreads();
    }

    // ---- epilogue: l lives in o[4] col 0 of each half-group; normalize ----
    #pragma unroll
    for (int r = 0; r < 8; ++r) {
        float l   = __shfl(o[4][r], hi * 16, 32);
        float inv = 1.0f / l;
        float* orow = O + ((size_t)b * L_ + (qbase + r + 8 * hi)) * (size_t)D_;
        #pragma unroll
        for (int dt = 0; dt < 4; ++dt)
            orow[dt * 16 + lo] = o[dt][r] * inv;
    }
}

extern "C" void kernel_launch(void* const* d_in, const int* in_sizes, int n_in,
                              void* d_out, int out_size, void* d_ws, size_t ws_size,
                              hipStream_t stream) {
    const float* Q  = (const float*)d_in[0];
    const float* K  = (const float*)d_in[1];
    const float* V  = (const float*)d_in[2];
    const float* sf = (const float*)d_in[3];
    float* O = (float*)d_out;

    dim3 grid(L_ / BM, B_, 1);   // 32 q-tiles x 16 batches = 512 workgroups
    dim3 block(256, 1, 1);       // 8 waves (wave32)
    hipLaunchKernelGGL(fa_fwd_kernel, grid, block, 0, stream, Q, K, V, sf, O);
}
